// StoryCrossAttention_78288663872512
// MI455X (gfx1250) — matrix operand order repaired and checked by reading, not compiled, not run on hardware
//
#include <hip/hip_runtime.h>
#include <hip/hip_bf16.h>
#include <math.h>

typedef __attribute__((ext_vector_type(16))) __bf16 v16bf;
typedef __attribute__((ext_vector_type(8)))  float  v8f;

#define L_SEQ 3072
#define DH    64
#define NHEAD 10
#define CDIM  640
#define MTOT  6144            // 2 * 3072 token rows
#define KT32  (CDIM / 32)     // 20 k-steps of 32
#define NT16  (CDIM / 16)     // 40 n-tiles of 16

__device__ __forceinline__ v8f zero8() {
  v8f z;
#pragma unroll
  for (int i = 0; i < 8; ++i) z[i] = 0.0f;
  return z;
}

// A-fragment (16x32 bf16): lane holds row (lane&15); elems 0..15 = K 2*kb+0..15,
// kb = (lane>>4)*8.
__device__ __forceinline__ v16bf a_frag(const __bf16* row, int kb) {
  v16bf r;
#pragma unroll
  for (int e = 0; e < 8; ++e) { r[e] = row[2 * kb + e]; r[e + 8] = row[2 * kb + 8 + e]; }
  return r;
}

__device__ __forceinline__ v8f wmma_bf16(v16bf a, v16bf b, v8f c) {
  return __builtin_amdgcn_wmma_f32_16x16x32_bf16(
      false, a, false, b, (short)0, c, false, false);
}

// CDNA5 async global->LDS copy (ASYNCcnt) and the split wait counters.
__device__ __forceinline__ void async_ld_b128(unsigned int lds, const void* g) {
  asm volatile("global_load_async_to_lds_b128 %0, %1, off"
               :: "v"(lds), "v"((unsigned long long)(uintptr_t)g) : "memory");
}
__device__ __forceinline__ void wait_async0() {
  asm volatile("s_wait_asynccnt 0x0" ::: "memory");
}
__device__ __forceinline__ void wait_ds0() {
  asm volatile("s_wait_dscnt 0x0" ::: "memory");
  __builtin_amdgcn_wave_barrier();
}
__device__ __forceinline__ unsigned int lds_off(const void* p) {
  return (unsigned int)(uintptr_t)p;  // generic AS3 ptr: low 32 bits = LDS addr
}

// ---------------------------------------------------------------- cvt f32->bf16
__global__ __launch_bounds__(256) void cvt_bf16(const float* __restrict__ src,
                                                __bf16* __restrict__ dst, int n) {
  int i = (blockIdx.x * blockDim.x + threadIdx.x) * 4;
  if (i + 3 < n) {
    float4 v = *(const float4*)(src + i);
    dst[i + 0] = (__bf16)v.x;
    dst[i + 1] = (__bf16)v.y;
    dst[i + 2] = (__bf16)v.z;
    dst[i + 3] = (__bf16)v.w;
  }
}

// ---------------------------------------------------- pack W into B-fragment order
// B-fragment (32x16 bf16): lane holds column n = nt*16 + (lane&15); element e
// = B[c0 + e][n], c0 = (lane>>4)*16. Layout: Wp[((kt*NT16 + nt)*32 + lane)*16 + e]
__global__ __launch_bounds__(256) void pack_weight(const float* __restrict__ W,
                                                   __bf16* __restrict__ Wp) {
  int t = blockIdx.x * blockDim.x + threadIdx.x;
  if (t >= KT32 * NT16 * 32) return;
  int lane = t & 31;
  int nt = (t >> 5) % NT16;
  int kt = t / (32 * NT16);
  int n  = nt * 16 + (lane & 15);
  int c0 = (lane >> 4) * 16;
  __bf16* dst = Wp + (size_t)t * 16;
#pragma unroll
  for (int e = 0; e < 16; ++e)
    dst[e] = (__bf16)W[(size_t)(kt * 32 + c0 + e) * CDIM + n];
}

// ------------------------------------------------------------------- QKV GEMM
// Block = 16 rows x 640 cols; 8 waves x 5 n-tiles. Base-pointer + constant-
// offset addressing so B loads use IOFFSET immediates (no u64 adds in loop).
__global__ __launch_bounds__(256) void qkv_gemm(const __bf16* __restrict__ Xbf,
                                                const __bf16* __restrict__ Wp,
                                                __bf16* __restrict__ Qbf,
                                                __bf16* __restrict__ Kbf,
                                                __bf16* __restrict__ Vt) {
  const int wave = threadIdx.x >> 5, lane = threadIdx.x & 31;
  const int r = lane & 15, kb = (lane >> 4) * 8, mlo = (lane >> 4) * 8;
  const int m0 = blockIdx.x * 16;
  const __bf16* arow = Xbf + (size_t)(m0 + r) * CDIM;
  for (int w = 0; w < 3; ++w) {
    const __bf16* bbase = Wp + (size_t)w * CDIM * CDIM +
                          (size_t)(wave * 5) * 512 + (size_t)lane * 16;
    v8f acc[5];
#pragma unroll
    for (int j = 0; j < 5; ++j) acc[j] = zero8();
    for (int kt = 0; kt < KT32; ++kt) {
      v16bf a = a_frag(arow + kt * 32, kb);
      const __bf16* bk = bbase + (size_t)kt * (NT16 * 512);
#pragma unroll
      for (int j = 0; j < 5; ++j) {
        v16bf b = *(const v16bf*)(bk + j * 512);
        acc[j] = wmma_bf16(a, b, acc[j]);
      }
    }
#pragma unroll
    for (int j = 0; j < 5; ++j) {
      int n = (wave * 5 + j) * 16 + r;
      int hh = n >> 6, d = n & 63;
#pragma unroll
      for (int i = 0; i < 8; ++i) {
        int m = m0 + mlo + i;
        int b = (m >= L_SEQ) ? 1 : 0;
        int l = m - b * L_SEQ;
        __bf16 val = (__bf16)acc[j][i];
        if (w == 0)
          Qbf[((size_t)(b * NHEAD + hh) * L_SEQ + l) * DH + d] = val;
        else if (w == 1)
          Kbf[((size_t)(b * NHEAD + hh) * L_SEQ + l) * DH + d] = val;
        else
          Vt[((size_t)(b * NHEAD + hh) * DH + d) * L_SEQ + l] = val;
      }
    }
  }
}

// -------------------------------------------------------------- flash attention
// Block = 4 waves x 16 queries. Per 64-key chunk: async-stage K (64x64) and
// V^T (64x64) into LDS ONCE per block, 8 WMMA for S, 32-lane f32 online
// softmax, 8 WMMA for P*V.
__global__ __launch_bounds__(128) void attn_flash(const __bf16* __restrict__ Qbf,
                                                  const __bf16* __restrict__ Kbf,
                                                  const __bf16* __restrict__ Vt,
                                                  const unsigned char* __restrict__ mask,
                                                  __bf16* __restrict__ Obf) {
  const int tid = threadIdx.x;
  const int wave = tid >> 5, lane = tid & 31;
  const int r = lane & 15, kb = (lane >> 4) * 8, c0 = (lane >> 4) * 16;
  const int mlo = (lane >> 4) * 8, half = lane >> 4;
  const int b = blockIdx.z, h = blockIdx.y;
  const int q0 = blockIdx.x * 64 + wave * 16;

  const __bf16* Qh = Qbf + (size_t)(b * NHEAD + h) * L_SEQ * DH;
  const __bf16* Kh = Kbf + (size_t)(b * NHEAD + h) * L_SEQ * DH;
  const __bf16* Vh = Vt + (size_t)(b * NHEAD + h) * DH * L_SEQ;
  // mask rows repeat per 1024-token image block; 16-row tiles never cross one.
  const unsigned char* mrow = mask + (size_t)q0 * L_SEQ;

  __shared__ __align__(32) __bf16 sK[64][64];   // keys x d      (8 KB)
  __shared__ __align__(32) __bf16 sV[64][64];   // d x keys      (8 KB)
  __shared__ float  sS[4][16][68];              // padded scores
  __shared__ __bf16 sP[4][16][68];
  __shared__ float  sM[4][16], sL[4][16], sA[4][16];
  __shared__ float  sMx[4][32], sPs[4][32];

  // async staging addresses: thread copies 4 K-segments + 4 V-segments of 16B
  const int srow = tid >> 3, sc8 = (tid & 7) * 8;
  unsigned int lK[4], lV[4];
#pragma unroll
  for (int t = 0; t < 4; ++t) {
    lK[t] = lds_off(&sK[srow + 16 * t][sc8]);
    lV[t] = lds_off(&sV[srow + 16 * t][sc8]);
  }
  const __bf16* gK0 = Kh + (size_t)srow * DH + sc8;
  const __bf16* gV0 = Vh + (size_t)srow * L_SEQ + sc8;

  v16bf aq0 = a_frag(Qh + (size_t)(q0 + r) * DH, kb);
  v16bf aq1 = a_frag(Qh + (size_t)(q0 + r) * DH + 32, kb);

  v8f acc[4];
#pragma unroll
  for (int j = 0; j < 4; ++j) acc[j] = zero8();
  if (lane < 16) { sM[wave][lane] = -3.0e38f; sL[wave][lane] = 0.0f; }
  __syncthreads();

  for (int kc = 0; kc < L_SEQ; kc += 64) {
    // ---- stage K/V chunk via CDNA5 async copy (once per block) -------------
#pragma unroll
    for (int t = 0; t < 4; ++t) {
      async_ld_b128(lK[t], gK0 + (size_t)t * 16 * DH + (size_t)kc * DH);
      async_ld_b128(lV[t], gV0 + (size_t)t * 16 * L_SEQ + kc);
    }
    wait_async0();
    __syncthreads();

    // ---- S = (Q K^T) * scale, masked, to LDS (wave-private tile) -----------
#pragma unroll
    for (int nt = 0; nt < 4; ++nt) {
      int key16 = nt * 16 + r;  // B column = key within chunk
      v16bf b0 = *(const v16bf*)(&sK[key16][c0]);
      v16bf b1 = *(const v16bf*)(&sK[key16][32 + c0]);
      v8f c = zero8();
      c = wmma_bf16(aq0, b0, c);
      c = wmma_bf16(aq1, b1, c);
      int ncol = nt * 16 + r;
      unsigned char keep = mrow[kc + ncol];
#pragma unroll
      for (int i = 0; i < 8; ++i)
        sS[wave][mlo + i][ncol] = keep ? c[i] * 0.125f : -1.0e9f;
    }
    wait_ds0();  // wave-local: sS tile is private to this wave

    // ---- 32-lane online softmax: lane = (row r, half) over 32 columns ------
    {
      int j0 = half * 32;
      float mOld = sM[wave][r];
      float lmax = -3.0e38f;
#pragma unroll 4
      for (int j = 0; j < 32; ++j) lmax = fmaxf(lmax, sS[wave][r][j0 + j]);
      sMx[wave][lane] = lmax;
      wait_ds0();
      float mNew = fmaxf(mOld, fmaxf(lmax, sMx[wave][lane ^ 16]));
      float alpha = __expf(mOld - mNew);
      float ps = 0.0f;
#pragma unroll 4
      for (int j = 0; j < 32; ++j) {
        float p = __expf(sS[wave][r][j0 + j] - mNew);
        sP[wave][r][j0 + j] = (__bf16)p;
        ps += p;
      }
      sPs[wave][lane] = ps;
      wait_ds0();
      if (half == 0) {
        float rs = ps + sPs[wave][lane ^ 16];
        sM[wave][r] = mNew;
        sL[wave][r] = sL[wave][r] * alpha + rs;
        sA[wave][r] = alpha;
      }
    }
    wait_ds0();

    // ---- rescale accumulators, then O += P * V ------------------------------
    float al[8];
#pragma unroll
    for (int i = 0; i < 8; ++i) al[i] = sA[wave][mlo + i];
#pragma unroll
    for (int nt = 0; nt < 4; ++nt)
#pragma unroll
      for (int i = 0; i < 8; ++i) acc[nt][i] *= al[i];
#pragma unroll
    for (int kk = 0; kk < 2; ++kk) {
      v16bf ap = a_frag(&sP[wave][r][kk * 32], kb);
#pragma unroll
      for (int nt = 0; nt < 4; ++nt) {
        v16bf bv = *(const v16bf*)(&sV[nt * 16 + r][kk * 32 + c0]);
        acc[nt] = wmma_bf16(ap, bv, acc[nt]);
      }
    }
    __syncthreads();  // protect sK/sV/sS before next chunk's staging
  }

  float linv[8];
#pragma unroll
  for (int i = 0; i < 8; ++i) linv[i] = 1.0f / sL[wave][mlo + i];
#pragma unroll
  for (int nt = 0; nt < 4; ++nt) {
#pragma unroll
    for (int i = 0; i < 8; ++i) {
      int q = q0 + mlo + i;
      int n = h * DH + nt * 16 + r;
      Obf[((size_t)(b * L_SEQ + q)) * CDIM + n] = (__bf16)(acc[nt][i] * linv[i]);
    }
  }
}

// ------------------------------------------------------------- output projection
__global__ __launch_bounds__(256) void out_gemm(const __bf16* __restrict__ Obf,
                                                const __bf16* __restrict__ Wp,
                                                const float* __restrict__ bout,
                                                float* __restrict__ out) {
  const int wave = threadIdx.x >> 5, lane = threadIdx.x & 31;
  const int r = lane & 15, kb = (lane >> 4) * 8, mlo = (lane >> 4) * 8;
  const int m0 = blockIdx.x * 16;
  const __bf16* arow = Obf + (size_t)(m0 + r) * CDIM;
  const __bf16* bbase = Wp + (size_t)(wave * 5) * 512 + (size_t)lane * 16;
  v8f acc[5];
#pragma unroll
  for (int j = 0; j < 5; ++j) acc[j] = zero8();
  for (int kt = 0; kt < KT32; ++kt) {
    v16bf a = a_frag(arow + kt * 32, kb);
    const __bf16* bk = bbase + (size_t)kt * (NT16 * 512);
#pragma unroll
    for (int j = 0; j < 5; ++j) {
      v16bf b = *(const v16bf*)(bk + j * 512);
      acc[j] = wmma_bf16(a, b, acc[j]);
    }
  }
#pragma unroll
  for (int j = 0; j < 5; ++j) {
    int n = (wave * 5 + j) * 16 + r;
    float bb = bout[n];
#pragma unroll
    for (int i = 0; i < 8; ++i) {
      int m = m0 + mlo + i;
      out[(size_t)m * CDIM + n] = acc[j][i] + bb;
    }
  }
}

// --------------------------------------------------------------------- launcher
extern "C" void kernel_launch(void* const* d_in, const int* in_sizes, int n_in,
                              void* d_out, int out_size, void* d_ws, size_t ws_size,
                              hipStream_t stream) {
  const float* hs = (const float*)d_in[0];
  const unsigned char* mask = (const unsigned char*)d_in[1];  // numpy bool, 1B
  const float* Wq = (const float*)d_in[2];
  const float* Wk = (const float*)d_in[3];
  const float* Wv = (const float*)d_in[4];
  const float* Wo = (const float*)d_in[5];
  const float* bo = (const float*)d_in[6];
  float* out = (float*)d_out;

  char* ws = (char*)d_ws;
  size_t off = 0;
  auto take = [&](size_t bytes) {
    char* p = ws + off;
    off += (bytes + 255) & ~(size_t)255;
    return p;
  };
  __bf16* Xbf = (__bf16*)take((size_t)MTOT * CDIM * 2);
  __bf16* Qbf = (__bf16*)take((size_t)2 * NHEAD * L_SEQ * DH * 2);
  __bf16* Kbf = (__bf16*)take((size_t)2 * NHEAD * L_SEQ * DH * 2);
  __bf16* Vt  = (__bf16*)take((size_t)2 * NHEAD * L_SEQ * DH * 2);
  __bf16* Obf = (__bf16*)take((size_t)MTOT * CDIM * 2);
  __bf16* Wp3 = (__bf16*)take((size_t)3 * CDIM * CDIM * 2);
  __bf16* Wpo = (__bf16*)take((size_t)CDIM * CDIM * 2);

  int n = MTOT * CDIM;
  cvt_bf16<<<n / 1024, 256, 0, stream>>>(hs, Xbf, n);

  int pk = KT32 * NT16 * 32;
  int pgrid = (pk + 255) / 256;
  pack_weight<<<pgrid, 256, 0, stream>>>(Wq, Wp3);
  pack_weight<<<pgrid, 256, 0, stream>>>(Wk, Wp3 + (size_t)CDIM * CDIM);
  pack_weight<<<pgrid, 256, 0, stream>>>(Wv, Wp3 + (size_t)2 * CDIM * CDIM);
  pack_weight<<<pgrid, 256, 0, stream>>>(Wo, Wpo);

  qkv_gemm<<<MTOT / 16, 256, 0, stream>>>(Xbf, Wp3, Qbf, Kbf, Vt);
  attn_flash<<<dim3(L_SEQ / 64, NHEAD, 2), 128, 0, stream>>>(Qbf, Kbf, Vt, mask, Obf);
  out_gemm<<<MTOT / 16, 256, 0, stream>>>(Obf, Wpo, bo, out);
}
